// StyledConv_3204045603354
// MI455X (gfx1250) — matrix operand order repaired and checked by reading, not compiled, not run on hardware
//
#include <hip/hip_runtime.h>
#include <math.h>

#define B_   8
#define H_   128
#define W_   128
#define IC_  256
#define OC_  256
#define STYLE_ 512
#define KTOT 2304                     // IC_ * 9
#define CONV_SCALE_ (1.0f / 48.0f)    // 1/sqrt(256*9)
#define LRELU_ 0.2f
#define ACT_SCALE_ 1.41421356237309515f
#define EPS_ 1e-8f

#define HALO_X 130                    // x' = 0..129 covers x = -1..128
#define NPIX (B_ * H_ * W_)           // 131072
#define NELEM (NPIX * IC_)            // 33,554,432

typedef __attribute__((ext_vector_type(16))) _Float16 v16h;
typedef __attribute__((ext_vector_type(8)))  _Float16 v8h;
typedef __attribute__((ext_vector_type(8)))  float    v8f;

// ---------------- Kernel 1: s = style @ mod_kernel + mod_bias  [B, IC] ----------------
__global__ __launch_bounds__(256) void style_linear_kernel(
    const float* __restrict__ style, const float* __restrict__ mod_kernel,
    const float* __restrict__ mod_bias, float* __restrict__ s) {
  int idx = blockIdx.x * 256 + threadIdx.x;     // 2048 total
  int b = idx >> 8;
  int ic = idx & 255;
  float acc = mod_bias[ic];
  for (int j = 0; j < STYLE_; ++j)
    acc = fmaf(style[b * STYLE_ + j], mod_kernel[j * IC_ + ic], acc);
  s[idx] = acc;
}

// ---------------- Kernel 2: modulate + demodulate, pack f16 weights ----------------
// Output layout: wt[b][oc][k], k = tap*256 + ic
__global__ __launch_bounds__(256) void modulate_kernel(
    const float* __restrict__ weight, const float* __restrict__ s,
    _Float16* __restrict__ wt) {
  __shared__ float red[256];
  int b  = blockIdx.x >> 8;
  int oc = blockIdx.x & 255;
  int ic = threadIdx.x;

  float m = CONV_SCALE_ * s[b * IC_ + ic];
  const float* wp = weight + (size_t)oc * KTOT + (size_t)ic * 9;
  float w[9];
  float ss = 0.0f;
#pragma unroll
  for (int t = 0; t < 9; ++t) {
    w[t] = wp[t] * m;
    ss = fmaf(w[t], w[t], ss);
  }
  red[ic] = ss;
  __syncthreads();
  for (int off = 128; off > 0; off >>= 1) {
    if (ic < off) red[ic] += red[ic + off];
    __syncthreads();
  }
  float demod = rsqrtf(red[0] + EPS_);
  _Float16* o = wt + (size_t)(b * OC_ + oc) * KTOT + ic;
#pragma unroll
  for (int t = 0; t < 9; ++t)
    o[t * IC_] = (_Float16)(w[t] * demod);
}

// ---------------- Kernel 3: input f32 -> f16 (NHWC preserved) ----------------
__global__ __launch_bounds__(256) void convert_input_kernel(
    const float* __restrict__ input, _Float16* __restrict__ inp_h) {
  size_t base = ((size_t)blockIdx.x * 256 + threadIdx.x) * 8;
  if (base >= (size_t)NELEM) return;
  const float4* p = (const float4*)(input + base);
  float4 f0 = p[0], f1 = p[1];
  v8h h;
  h[0] = (_Float16)f0.x; h[1] = (_Float16)f0.y; h[2] = (_Float16)f0.z; h[3] = (_Float16)f0.w;
  h[4] = (_Float16)f1.x; h[5] = (_Float16)f1.y; h[6] = (_Float16)f1.z; h[7] = (_Float16)f1.w;
  *(v8h*)(inp_h + base) = h;
}

// ---------------- Kernel 4: implicit-GEMM conv + epilogue (WMMA f16) ----------------
// Block: 256 threads (8 waves). Tile: batch b, one full row y (128 px) x all 256 OC.
// Wave w -> oc0 = 32*w; per wave: 8 M-tiles x 2 N-tiles of 16x16, K = 2304.
// A halo (3 x 130 x 256 f16 = 195KB) staged into dynamic LDS with
// global_load_async_to_lds_b128 (ASYNCcnt); borders zero-filled with ds stores.
__global__ __launch_bounds__(256, 1) void conv_wmma_kernel(
    const _Float16* __restrict__ inp_h, const _Float16* __restrict__ wt,
    const float* __restrict__ noise, const float* __restrict__ nw_p,
    const float* __restrict__ act_bias, float* __restrict__ out) {
  extern __shared__ _Float16 lds[];   // 3 * 130 * 256 halfs = 199,680 bytes

  int bid = blockIdx.x;               // 0..1023
  int y = bid & 127;
  int b = bid >> 7;
  int tid = threadIdx.x;

  // ---- Stage halo: rows y-1..y+1, x = -1..128, all 256 ic ----
  // 3 * 130 * 32 chunks of 8 halfs (16B each)
  for (int t = tid; t < 3 * HALO_X * 32; t += 256) {
    int r   = t / (HALO_X * 32);
    int rem = t - r * (HALO_X * 32);
    int xp  = rem >> 5;               // 0..129
    int icc = (rem & 31) << 3;        // ic chunk of 8
    int yy = y - 1 + r;
    int xx = xp - 1;
    int ldsoff = (((r * HALO_X + xp) << 8) + icc) * 2;   // byte offset in LDS
    if ((unsigned)yy < (unsigned)H_ && (unsigned)xx < (unsigned)W_) {
      const _Float16* g = inp_h + (((size_t)((b * H_ + yy) * W_ + xx)) << 8) + icc;
      unsigned long long ga = (unsigned long long)(uintptr_t)g;
      asm volatile("global_load_async_to_lds_b128 %0, %1, off"
                   :: "v"(ldsoff), "v"(ga) : "memory");
    } else {
      v8h z = {};
      *(v8h*)(lds + ((r * HALO_X + xp) << 8) + icc) = z;
    }
  }
  asm volatile("s_wait_asynccnt 0x0" ::: "memory");
  __syncthreads();

  int lane = tid & 31;
  int wave = tid >> 5;
  int ln = lane & 15;      // M (for A) / N (for B/D)
  int hi = lane >> 4;      // lane-group select
  int oc0 = wave << 5;

  v8f acc[8][2] = {};
  const _Float16* wb = wt + (size_t)b * OC_ * KTOT;

  for (int tap = 0; tap < 9; ++tap) {
    int r  = tap / 3;                 // dy + 1
    int dx = tap - r * 3 - 1;
    // A: lds element offset for pixel m = mt*16 + ln, shifted by dx (halo +1)
    int arow = (r * HALO_X + (ln + dx + 1)) << 8;
    // B: elems 0..7 = K hi*8+0..7, 8..15 = K 16+hi*8+0..7
    const _Float16* b0 = wb + (size_t)(oc0 + ln) * KTOT + tap * IC_ + hi * 8;
    const _Float16* b1 = b0 + (size_t)16 * KTOT;

#pragma unroll 2
    for (int icc = 0; icc < 8; ++icc) {
      int ic0 = icc << 5;
      v8h bl0 = *(const v8h*)(b0 + ic0);
      v8h bh0 = *(const v8h*)(b0 + ic0 + 16);
      v8h bl1 = *(const v8h*)(b1 + ic0);
      v8h bh1 = *(const v8h*)(b1 + ic0 + 16);
      v16h bf0 = __builtin_shufflevector(bl0, bh0, 0,1,2,3,4,5,6,7,8,9,10,11,12,13,14,15);
      v16h bf1 = __builtin_shufflevector(bl1, bh1, 0,1,2,3,4,5,6,7,8,9,10,11,12,13,14,15);
      int ao = arow + ic0 + hi * 8;   // A: elems 0..7 = K hi*8+0..7, 8..15 = K 16+hi*8+0..7
#pragma unroll
      for (int mt = 0; mt < 8; ++mt) {
        v8h al = *(const v8h*)(&lds[ao + mt * (16 << 8)]);
        v8h ah = *(const v8h*)(&lds[ao + mt * (16 << 8) + 16]);
        v16h a = __builtin_shufflevector(al, ah, 0,1,2,3,4,5,6,7,8,9,10,11,12,13,14,15);
        acc[mt][0] = __builtin_amdgcn_wmma_f32_16x16x32_f16(false, a, false, bf0, (short)0, acc[mt][0], false, false);
        acc[mt][1] = __builtin_amdgcn_wmma_f32_16x16x32_f16(false, a, false, bf1, (short)0, acc[mt][1], false, false);
      }
    }
  }

  // ---- Epilogue: + noise*nw + act_bias, leaky_relu * sqrt(2) ----
  float nw  = nw_p[0];
  float ab0 = act_bias[oc0 + ln];
  float ab1 = act_bias[oc0 + 16 + ln];
  size_t obase = ((size_t)(b * H_ + y) * W_) * OC_;
  const float* nrow = noise + (size_t)(b * H_ + y) * W_;

#pragma unroll
  for (int mt = 0; mt < 8; ++mt) {
#pragma unroll
    for (int rr = 0; rr < 8; ++rr) {
      int x = mt * 16 + hi * 8 + rr;        // D layout: M = rr + 8*hi
      float nz = nw * nrow[x];
      float v0 = acc[mt][0][rr] + nz + ab0;
      v0 = (v0 < 0.f ? v0 * LRELU_ : v0) * ACT_SCALE_;
      out[obase + (size_t)x * OC_ + oc0 + ln] = v0;
      float v1 = acc[mt][1][rr] + nz + ab1;
      v1 = (v1 < 0.f ? v1 * LRELU_ : v1) * ACT_SCALE_;
      out[obase + (size_t)x * OC_ + oc0 + 16 + ln] = v1;
    }
  }
}

extern "C" void kernel_launch(void* const* d_in, const int* in_sizes, int n_in,
                              void* d_out, int out_size, void* d_ws, size_t ws_size,
                              hipStream_t stream) {
  const float* input      = (const float*)d_in[0];
  const float* style      = (const float*)d_in[1];
  const float* noise      = (const float*)d_in[2];
  const float* weight     = (const float*)d_in[3];
  const float* mod_kernel = (const float*)d_in[4];
  const float* mod_bias   = (const float*)d_in[5];
  const float* noise_w    = (const float*)d_in[6];
  const float* act_bias   = (const float*)d_in[7];
  float* out = (float*)d_out;

  // workspace layout:
  //   [0, 8KB)            s : [8, 256] f32
  //   [8KB, +9.44MB)      wt : [8][256][2304] f16 (modulated+demodulated weights)
  //   [+, +67.1MB)        inp_h : input converted to f16 (NHWC)
  float* s_ws = (float*)d_ws;
  _Float16* wt = (_Float16*)((char*)d_ws + 8192);
  _Float16* inp_h = (_Float16*)((char*)d_ws + 8192 + (size_t)B_ * OC_ * KTOT * 2);

  style_linear_kernel<<<B_ * IC_ / 256, 256, 0, stream>>>(style, mod_kernel, mod_bias, s_ws);
  modulate_kernel<<<B_ * OC_, 256, 0, stream>>>(weight, s_ws, wt);
  convert_input_kernel<<<NELEM / (256 * 8), 256, 0, stream>>>(input, inp_h);

  size_t lds_bytes = (size_t)3 * HALO_X * IC_ * sizeof(_Float16);   // 199,680 B
  conv_wmma_kernel<<<B_ * H_, 256, lds_bytes, stream>>>(inp_h, wt, noise, noise_w, act_bias, out);
}
